// MultiHeadAttention_21363167330837
// MI455X (gfx1250) — matrix operand order repaired and checked by reading, not compiled, not run on hardware
//
#include <hip/hip_runtime.h>
#include <hip/hip_bf16.h>

// ---------------------------------------------------------------------------
// MHA (causal, floor-scaled scores) + output projection for MI455X (gfx1250).
// f32 inputs converted to f16 WMMA operands, f32 accumulation.
// Kernel 1: flash attention, one wave per (b, h, 16-query tile) -> f16 in d_ws
// Kernel 2: WMMA GEMM Y = A @ W^T + b; A tile staged via TENSOR_LOAD_TO_LDS.
// ---------------------------------------------------------------------------

typedef _Float16 half4  __attribute__((ext_vector_type(4)));
typedef _Float16 half8  __attribute__((ext_vector_type(8)));
typedef _Float16 v16h   __attribute__((ext_vector_type(16)));
typedef float    v8f    __attribute__((ext_vector_type(8)));
typedef unsigned u32x4  __attribute__((ext_vector_type(4)));
typedef int      i32x4  __attribute__((ext_vector_type(4)));
typedef int      i32x8  __attribute__((ext_vector_type(8)));

#define N_HEADS 16
#define HD      128
#define TLEN    2048
#define CDIM    2048
#define BATCH   4

#define QS_STR  136   // halves per row (16B-aligned stride, padded)
#define KS_STR  136
#define VT_STR  40
#define PT_STR  40
#define AS_STR  32    // TDM writes the A tile contiguously (64 x 32 halves)
#define BS_STR  40

// WMMA f32 <= f16 x f16, 16x16x32 (codegen-confirmed signature)
__device__ __forceinline__ v8f wmma16(v16h a, v16h b, v8f c) {
  return __builtin_amdgcn_wmma_f32_16x16x32_f16(
      /*neg_a=*/false, a, /*neg_b=*/false, b,
      /*c_mod=*/(short)0, c, /*reuse_a=*/false, /*reuse_b=*/false);
}

// A-matrix 16x32 f16 fragment: per lane-half, elements are
// K = {kb..kb+15} (contiguous). Caller passes row pointer already offset by kb.
__device__ __forceinline__ v16h frag_a(const _Float16* p) {
  half8 lo = *(const half8*)(p);
  half8 hi = *(const half8*)(p + 8);
  return __builtin_shufflevector(lo, hi, 0,1,2,3,4,5,6,7,8,9,10,11,12,13,14,15);
}

// B-matrix 32x16 f16 fragment: per lane-half, 16 contiguous K values.
__device__ __forceinline__ v16h frag_b(const _Float16* p) {
  half8 lo = *(const half8*)(p);
  half8 hi = *(const half8*)(p + 8);
  return __builtin_shufflevector(lo, hi, 0,1,2,3,4,5,6,7,8,9,10,11,12,13,14,15);
}

// 16-lane reductions via ds_swizzle XOR patterns (wave32-safe).
#define DS_SWIZF(x, pat) \
  __int_as_float(__builtin_amdgcn_ds_swizzle(__float_as_int(x), (pat)))

__device__ __forceinline__ float redmax16(float v) {
  v = fmaxf(v, DS_SWIZF(v, 0x041f));  // SWAPX1
  v = fmaxf(v, DS_SWIZF(v, 0x081f));  // SWAPX2
  v = fmaxf(v, DS_SWIZF(v, 0x101f));  // SWAPX4
  v = fmaxf(v, DS_SWIZF(v, 0x201f));  // SWAPX8
  return v;
}
__device__ __forceinline__ float redsum16(float v) {
  v += DS_SWIZF(v, 0x041f);
  v += DS_SWIZF(v, 0x081f);
  v += DS_SWIZF(v, 0x101f);
  v += DS_SWIZF(v, 0x201f);
  return v;
}

// ---------------------------------------------------------------------------
// Kernel 1: flash attention. grid = (T/16, H, B), block = 32 (one wave).
// ---------------------------------------------------------------------------
__global__ __launch_bounds__(32)
void attn_kernel(const float* __restrict__ vg, const float* __restrict__ kg,
                 const float* __restrict__ qg, _Float16* __restrict__ outh) {
  __shared__ __align__(16) _Float16 qs[16 * QS_STR];   // Q tile  [q][d]
  __shared__ __align__(16) _Float16 ks[32 * KS_STR];   // K tile  [key][d]
  __shared__ __align__(16) _Float16 vt[HD * VT_STR];   // V tile  [d][key] (T)
  __shared__ __align__(16) _Float16 pt[16 * PT_STR];   // P tile  [q][key]

  const int lane  = threadIdx.x;     // 0..31
  const int q0    = blockIdx.x * 16;
  const int h     = blockIdx.y;
  const int b     = blockIdx.z;
  const int n     = lane & 15;
  const int half  = lane >> 4;       // 0/1
  const int kbA   = half * 16;       // A-fragment K base
  const int khB   = half * 16;       // B-fragment K base

  const size_t base = (size_t)b * TLEN * CDIM + (size_t)h * HD;

  // ---- stage Q tile (16x128 f32 -> f16), 512 float4 chunks over 32 lanes ----
  #pragma unroll
  for (int i = 0; i < 16; ++i) {
    int c = lane + 32 * i;
    int row = c >> 5;
    int off = (c & 31) * 4;
    const float4 f = *(const float4*)(qg + base + (size_t)(q0 + row) * CDIM + off);
    half4 hv = {(_Float16)f.x, (_Float16)f.y, (_Float16)f.z, (_Float16)f.w};
    *(half4*)(qs + row * QS_STR + off) = hv;
  }
  asm volatile("s_wait_dscnt 0" ::: "memory");

  // ---- Q A-fragments: 4 d-blocks of 32 ----
  v16h aq[4];
  #pragma unroll
  for (int db = 0; db < 4; ++db)
    aq[db] = frag_a(qs + n * QS_STR + db * 32 + kbA);

  v8f acc[8];
  #pragma unroll
  for (int db = 0; db < 8; ++db) acc[db] = (v8f){};
  float mrow[8], lrow[8];
  #pragma unroll
  for (int j = 0; j < 8; ++j) { mrow[j] = -__builtin_inff(); lrow[j] = 0.f; }

  const float rscale = 0.08838834764831845f;  // 1/sqrt(128)

  for (int kt = 0; kt < q0 + 16; kt += 32) {
    // prefetch next K/V tile rows (global_prefetch_b8)
    if (kt + 32 < q0 + 16) {
      __builtin_prefetch(kg + base + (size_t)(kt + 32 + lane) * CDIM, 0, 3);
      __builtin_prefetch(vg + base + (size_t)(kt + 32 + lane) * CDIM, 0, 3);
    }

    // ---- stage K (row-major) and V (transposed) tiles, one row per lane ----
    const float* krow = kg + base + (size_t)(kt + lane) * CDIM;
    const float* vrow = vg + base + (size_t)(kt + lane) * CDIM;
    #pragma unroll
    for (int i = 0; i < 32; ++i) {
      const float4 f = *(const float4*)(krow + i * 4);
      half4 hv = {(_Float16)f.x, (_Float16)f.y, (_Float16)f.z, (_Float16)f.w};
      *(half4*)(ks + lane * KS_STR + i * 4) = hv;
    }
    #pragma unroll
    for (int i = 0; i < 32; ++i) {
      const float4 f = *(const float4*)(vrow + i * 4);
      vt[(i * 4 + 0) * VT_STR + lane] = (_Float16)f.x;
      vt[(i * 4 + 1) * VT_STR + lane] = (_Float16)f.y;
      vt[(i * 4 + 2) * VT_STR + lane] = (_Float16)f.z;
      vt[(i * 4 + 3) * VT_STR + lane] = (_Float16)f.w;
    }
    asm volatile("s_wait_dscnt 0" ::: "memory");

    // ---- S = Q K^T for 2 key-halves of 16, scale -> floor -> causal mask ----
    v8f s[2];
    #pragma unroll
    for (int nh = 0; nh < 2; ++nh) {
      v8f c = (v8f){};
      const _Float16* kp = ks + (nh * 16 + n) * KS_STR;  // B: col = key
      #pragma unroll
      for (int db = 0; db < 4; ++db)
        c = wmma16(aq[db], frag_b(kp + db * 32 + khB), c);
      #pragma unroll
      for (int j = 0; j < 8; ++j) {
        float sv = floorf(c[j] * rscale);
        int col = kt + nh * 16 + n;
        int row = q0 + j + half * 8;
        s[nh][j] = (col <= row) ? sv : -__builtin_inff();
      }
    }

    // ---- online softmax update + stash P (f16) into LDS ----
    #pragma unroll
    for (int j = 0; j < 8; ++j) {
      float tm   = redmax16(fmaxf(s[0][j], s[1][j]));
      float mnew = fmaxf(mrow[j], tm);
      float alpha = __expf(mrow[j] - mnew);
      mrow[j] = mnew;
      float p0 = __expf(s[0][j] - mnew);
      float p1 = __expf(s[1][j] - mnew);
      lrow[j] = lrow[j] * alpha + redsum16(p0 + p1);
      #pragma unroll
      for (int db = 0; db < 8; ++db) acc[db][j] *= alpha;
      pt[(j + half * 8) * PT_STR + n]      = (_Float16)p0;
      pt[(j + half * 8) * PT_STR + 16 + n] = (_Float16)p1;
    }
    asm volatile("s_wait_dscnt 0" ::: "memory");

    // ---- O += P (16x32) x V (32x128) ----
    v16h pa = frag_a(pt + n * PT_STR + kbA);
    #pragma unroll
    for (int db = 0; db < 8; ++db) {
      v16h bv = frag_b(vt + (db * 16 + n) * VT_STR + khB);  // B: col = d
      acc[db] = wmma16(pa, bv, acc[db]);
    }
  }

  // ---- epilogue: normalize and store f16 attention output to workspace ----
  #pragma unroll
  for (int j = 0; j < 8; ++j) {
    float rl = 1.0f / lrow[j];
    size_t ro = base + (size_t)(q0 + j + half * 8) * CDIM;
    #pragma unroll
    for (int db = 0; db < 8; ++db)
      outh[ro + db * 16 + n] = (_Float16)(acc[db][j] * rl);
  }
}

// ---------------------------------------------------------------------------
// Kernel 2: Y[m][n] = sum_k A[m][k] * W[n][k] + b[n].
// grid = (M/64, N/64), block = 128 (4 waves); each wave does a 16x64 strip.
// A tile (64x32 f16, raw copy) is staged with the Tensor Data Mover.
// ---------------------------------------------------------------------------
__global__ __launch_bounds__(128)
void proj_kernel(const _Float16* __restrict__ A, const float* __restrict__ W,
                 const float* __restrict__ bias, float* __restrict__ Y) {
  __shared__ __align__(16) _Float16 as[64 * AS_STR];  // [m][k], TDM-filled
  __shared__ __align__(16) _Float16 bs[64 * BS_STR];  // [n][k]

  const int tid  = threadIdx.x;
  const int lane = tid & 31;
  const int w    = tid >> 5;
  const int n    = lane & 15;
  const int half = lane >> 4;
  const int kbA  = half * 16;
  const int khB  = half * 16;
  const int m0   = blockIdx.x * 64;
  const int n0   = blockIdx.y * 64;

  const unsigned lds_as = (unsigned)(size_t)(void*)as;  // LDS byte offset

  v8f acc[4];
  #pragma unroll
  for (int nb = 0; nb < 4; ++nb) acc[nb] = (v8f){};

  for (int k0 = 0; k0 < CDIM; k0 += 32) {
    // ---- stage A via Tensor Data Mover: 64x32 f16 tile from workspace ----
    if (w == 0) {  // wave-uniform branch; TDM ignores EXEC, issue once
      unsigned long long ga =
          (unsigned long long)(const void*)(A + (size_t)m0 * CDIM + k0);
      // D# group0: count=1 | lds_addr | global_addr(57b) | type=2 (bits 127:126)
      u32x4 g0 = { 1u,
                   lds_as,
                   (unsigned)(ga & 0xFFFFFFFFu),
                   (unsigned)(((ga >> 32) & 0x01FFFFFFu) | 0x80000000u) };
      // D# group1: data_size=2B (code 1, bits 17:16); tensor_dim0=2048
      // (bits 79:48); tensor_dim1=8192 (bits 111:80); tile_dim0=32
      // (bits 127:112); tile_dim1=64 (bits 143:128); dim0_stride=2048
      // (bits 207:160); no padding/iterate/multicast.
      i32x8 g1 = { 0x00010000,          // data_size=1 (2 bytes)
                   0x08000000,          // tensor_dim0[15:0]=2048 << 16
                   0x20000000,          // tensor_dim1[15:0]=8192 << 16
                   0x00200000,          // tile_dim0=32 << 16
                   0x00000040,          // tile_dim1=64
                   0x00000800,          // tensor_dim0_stride=2048
                   0x00000000,
                   0x00000000 };
      i32x4 z4 = { 0, 0, 0, 0 };        // 2D tensor: groups 2/3 disabled
      i32x8 z8 = { 0, 0, 0, 0, 0, 0, 0, 0 };
      __builtin_amdgcn_tensor_load_to_lds(g0, g1, z4, z4, z8, 0);
      __builtin_amdgcn_s_wait_tensorcnt(0);
    }

    // ---- stage W: 64x32 f32 = 512 float4 chunks / 128 threads -> f16 ----
    #pragma unroll
    for (int i = 0; i < 4; ++i) {
      int c = tid + 128 * i;
      int row = c >> 3, off = (c & 7) * 4;
      const float4 f = *(const float4*)(W + (size_t)(n0 + row) * CDIM + k0 + off);
      half4 hv = {(_Float16)f.x, (_Float16)f.y, (_Float16)f.z, (_Float16)f.w};
      *(half4*)(bs + row * BS_STR + off) = hv;
    }
    __syncthreads();

    v16h av = frag_a(as + (w * 16 + n) * AS_STR + kbA);
    #pragma unroll
    for (int nb = 0; nb < 4; ++nb) {
      v16h bv = frag_b(bs + (nb * 16 + n) * BS_STR + khB);
      acc[nb] = wmma16(av, bv, acc[nb]);
    }
    __syncthreads();
  }

  #pragma unroll
  for (int nb = 0; nb < 4; ++nb) {
    float bb = bias[n0 + nb * 16 + n];
    #pragma unroll
    for (int j = 0; j < 8; ++j)
      Y[(size_t)(m0 + w * 16 + j + half * 8) * CDIM + n0 + nb * 16 + n] =
          acc[nb][j] + bb;
  }
}

// ---------------------------------------------------------------------------
extern "C" void kernel_launch(void* const* d_in, const int* in_sizes, int n_in,
                              void* d_out, int out_size, void* d_ws, size_t ws_size,
                              hipStream_t stream) {
  (void)in_sizes; (void)n_in; (void)out_size; (void)ws_size;
  const float* v    = (const float*)d_in[0];
  const float* k    = (const float*)d_in[1];
  const float* q    = (const float*)d_in[2];
  const float* Wout = (const float*)d_in[3];
  const float* bout = (const float*)d_in[4];
  float*       out  = (float*)d_out;
  _Float16*    ws   = (_Float16*)d_ws;   // (B*T) x C f16 attention output

  dim3 g1(TLEN / 16, N_HEADS, BATCH);
  attn_kernel<<<g1, 32, 0, stream>>>(v, k, q, ws);

  dim3 g2((BATCH * TLEN) / 64, CDIM / 64);
  proj_kernel<<<g2, 128, 0, stream>>>(ws, Wout, bout, out);
}
